// Contrast_54631984005844
// MI455X (gfx1250) — compile-verified, matrix operand order checked by reading, not run on hardware
//
#include <hip/hip_runtime.h>
#include <hip/hip_bf16.h>
#include <math.h>

// ---------------- problem constants ----------------
#define TAU_F   0.8f
#define LAM_F   0.5f
#define NROWS   4096
#define HDIM    512

// ---------------- GEMM tiling ----------------
#define BM   128
#define BN   128
#define BK   32
#define LDSS 40              // bf16 elems per LDS row (32 + 8 pad) -> 80B rows, 16B-aligned chunks
#define BUF_USH (BM * LDSS)  // 5120 ushorts per operand buffer

typedef __attribute__((ext_vector_type(8)))  float  v8f;
typedef __attribute__((ext_vector_type(16))) __bf16 v16bf;

// f32 -> bf16 (round to nearest even), raw ushort
__device__ __forceinline__ unsigned short f2bf(float f) {
  unsigned u = __builtin_bit_cast(unsigned, f);
  u += 0x7FFFu + ((u >> 16) & 1u);
  return (unsigned short)(u >> 16);
}
__device__ __forceinline__ float bf2f(unsigned short h) {
  unsigned u = ((unsigned)h) << 16;
  return __builtin_bit_cast(float, u);
}

union FragU { uint4 q[2]; v16bf v; };

// lane fragment = two contiguous 8-bf16 chunks (K=kb..kb+7, kb+16..kb+23)
__device__ __forceinline__ v16bf load_frag(const unsigned short* p0, const unsigned short* p1) {
  FragU u;
  u.q[0] = *(const uint4*)p0;
  u.q[1] = *(const uint4*)p1;
  return u.v;
}

// Async-copy one 128x32 bf16 tile (global, row-major, leading dim ld) into an LDS
// operand buffer. 512 16B chunks / 256 threads = 2 per thread. Tracked by ASYNCcnt.
__device__ __forceinline__ void async_copy_tile(
    const unsigned short* __restrict__ G, int ld, int R0, int kt,
    unsigned short* sdst, int tid)
{
#pragma unroll
  for (int it = 0; it < 2; ++it) {
    const int q   = it * 256 + tid;
    const int row = q >> 2;
    const int c   = (q & 3) << 3;                          // bf16 column within slab
    const unsigned lds = (unsigned)(uintptr_t)(sdst + row * LDSS + c);
    const unsigned long long ga =
        (unsigned long long)(uintptr_t)(G + (size_t)(R0 + row) * ld + kt + c);
    asm volatile("global_load_async_to_lds_b128 %0, %1, off"
                 :: "v"(lds), "v"(ga) : "memory");
  }
}

// One 32-deep K-slab of WMMAs from a fixed pair of LDS buffers.
__device__ __forceinline__ void mma_slab(
    const unsigned short* As, const unsigned short* Bs,
    int mrow, int kb, int nsel, v8f acc[8])
{
  const unsigned short* ap = As + mrow * LDSS + kb;
  v16bf a = load_frag(ap, ap + 16);
  v16bf bfrag[8];
#pragma unroll
  for (int t = 0; t < 8; ++t) {
    const unsigned short* bp = Bs + ((t << 4) + nsel) * LDSS + kb;
    bfrag[t] = load_frag(bp, bp + 16);
  }
#pragma unroll
  for (int t = 0; t < 8; ++t)
    acc[t] = __builtin_amdgcn_wmma_f32_16x16x32_bf16(
        false, a, false, bfrag[t], (short)0, acc[t], false, false);
}

// C[I0:+128, J0:+128] += A[MxK] * B[NxK]^T, bf16 operands in global, f32 accum.
// 256 threads = 8 waves; wave w owns rows I0+16w..+15, 8 WMMA tiles across 128 cols.
// Statically double-buffered async global->LDS staging overlapped with WMMA
// (K-loop unrolled by 2 so buffer choice is compile-time -> ds_load path survives).
__device__ __forceinline__ void gemm_core_bf16(
    const unsigned short* __restrict__ A, const unsigned short* __restrict__ B,
    int lda, int ldb, int K, int I0, int J0,
    unsigned short* smem /* 4 * BUF_USH */, v8f acc[8])
{
  const int tid  = threadIdx.x;
  const int w    = tid >> 5;
  const int lane = tid & 31;
  const int mrow = (w << 4) + (lane & 15);
  const int nsel = lane & 15;
  const int kb   = (lane >> 4) << 3;

  unsigned short* As0 = smem;
  unsigned short* Bs0 = smem + BUF_USH;
  unsigned short* As1 = smem + 2 * BUF_USH;
  unsigned short* Bs1 = smem + 3 * BUF_USH;

  async_copy_tile(A, lda, I0, 0, As0, tid);
  async_copy_tile(B, ldb, J0, 0, Bs0, tid);

  for (int kt = 0; kt < K; kt += 2 * BK) {
    // ---- slab kt in buf0 ----
    asm volatile("s_wait_asynccnt 0x0" ::: "memory");      // copies into buf0 landed
    __syncthreads();                                       // ... for every wave; buf1 free
    if (kt + BK < K) {
      async_copy_tile(A, lda, I0, kt + BK, As1, tid);
      async_copy_tile(B, ldb, J0, kt + BK, Bs1, tid);
    }
    mma_slab(As0, Bs0, mrow, kb, nsel, acc);

    // ---- slab kt+BK in buf1 ----
    asm volatile("s_wait_asynccnt 0x0" ::: "memory");      // copies into buf1 landed
    __syncthreads();                                       // ... for every wave; buf0 free
    if (kt + 2 * BK < K) {
      async_copy_tile(A, lda, I0, kt + 2 * BK, As0, tid);
      async_copy_tile(B, ldb, J0, kt + 2 * BK, Bs0, tid);
    }
    if (kt + BK < K)
      mma_slab(As1, Bs1, mrow, kb, nsel, acc);
  }
}

// ---------------- elementwise f32 -> bf16 ----------------
__global__ __launch_bounds__(256)
void cvt_f32_bf16_kernel(const float* __restrict__ src, unsigned short* __restrict__ dst, int n)
{
  const int i = (blockIdx.x * blockDim.x + threadIdx.x) << 2;
  if (i < n) {
    const float4 v = *(const float4*)(src + i);
    ushort4 o;
    o.x = f2bf(v.x); o.y = f2bf(v.y); o.z = f2bf(v.z); o.w = f2bf(v.w);
    *(ushort4*)(dst + i) = o;
  }
}

// ---------------- projection GEMM: out_bf16 = act(A @ W^T + bias) ----------------
__global__ __launch_bounds__(256)
void gemm_bias_act_kernel(const unsigned short* __restrict__ A, const unsigned short* __restrict__ W,
                          const float* __restrict__ bias, unsigned short* __restrict__ out,
                          int M, int N, int K, int apply_elu)
{
  __shared__ __align__(16) unsigned short smem[4 * BUF_USH];
  const int I0 = blockIdx.y * BM;
  const int J0 = blockIdx.x * BN;

  v8f acc[8];
  const v8f vz = {0.f,0.f,0.f,0.f,0.f,0.f,0.f,0.f};
#pragma unroll
  for (int t = 0; t < 8; ++t) acc[t] = vz;

  gemm_core_bf16(A, W, K, K, K, I0, J0, smem, acc);

  const int w = threadIdx.x >> 5, lane = threadIdx.x & 31;
  const int nbase = lane & 15, rhalf = (lane >> 4) << 3;
#pragma unroll
  for (int t = 0; t < 8; ++t) {
    const int n = J0 + (t << 4) + nbase;
    const float bv = bias[n];
#pragma unroll
    for (int r = 0; r < 8; ++r) {
      const int i = I0 + (w << 4) + rhalf + r;
      float v = acc[t][r] + bv;
      if (apply_elu) v = (v > 0.f) ? v : expm1f(v);
      out[(size_t)i * N + n] = f2bf(v);
    }
  }
}

// ---- similarity: m = exp((zm zs^T)/(|zm||zs| TAU)); fused row/col (masked) sums ----
__global__ __launch_bounds__(256)
void sim_reduce_kernel(const unsigned short* __restrict__ zm, const unsigned short* __restrict__ zs,
                       const float* __restrict__ nm, const float* __restrict__ ns,
                       const float* __restrict__ pos,
                       float* __restrict__ rowsum, float* __restrict__ rowmask,
                       float* __restrict__ colsum, float* __restrict__ colmask,
                       int Nfull, int K)
{
  __shared__ __align__(16) unsigned short smem[4 * BUF_USH];
  __shared__ float srow[BM], srowm[BM], scol[BN], scolm[BN];
  const int tid = threadIdx.x;
  if (tid < BM) { srow[tid] = 0.f; srowm[tid] = 0.f; scol[tid] = 0.f; scolm[tid] = 0.f; }

  const int I0 = blockIdx.y * BM;
  const int J0 = blockIdx.x * BN;

  v8f acc[8];
  const v8f vz = {0.f,0.f,0.f,0.f,0.f,0.f,0.f,0.f};
#pragma unroll
  for (int t = 0; t < 8; ++t) acc[t] = vz;

  gemm_core_bf16(zm, zs, K, K, K, I0, J0, smem, acc);

  const int w = tid >> 5, lane = tid & 31;
  const int nbase = lane & 15, rhalf = (lane >> 4) << 3;

  float nsv[8];
#pragma unroll
  for (int t = 0; t < 8; ++t) nsv[t] = ns[J0 + (t << 4) + nbase];

  float rs[8] = {0,0,0,0,0,0,0,0}, rm[8] = {0,0,0,0,0,0,0,0};
  float cs[8] = {0,0,0,0,0,0,0,0}, cm[8] = {0,0,0,0,0,0,0,0};
#pragma unroll
  for (int r = 0; r < 8; ++r) {
    const int i = I0 + (w << 4) + rhalf + r;
    const float nmi = nm[i];
#pragma unroll
    for (int t = 0; t < 8; ++t) {
      const int j = J0 + (t << 4) + nbase;
      const float s = __expf(acc[t][r] / (nmi * nsv[t] * TAU_F));
      rs[r] += s;
      cs[t] += s;
      rm[r] += s * pos[(size_t)i * Nfull + j];   // sum_j m[i,j]*pos[i,j]
      cm[t] += s * pos[(size_t)j * Nfull + i];   // sum_i m[i,j]*pos[j,i]
    }
  }
  __syncthreads();
#pragma unroll
  for (int r = 0; r < 8; ++r) {
    const int loc = (w << 4) + rhalf + r;
    atomicAdd(&srow[loc],  rs[r]);
    atomicAdd(&srowm[loc], rm[r]);
  }
#pragma unroll
  for (int t = 0; t < 8; ++t) {
    const int loc = (t << 4) + nbase;
    atomicAdd(&scol[loc],  cs[t]);
    atomicAdd(&scolm[loc], cm[t]);
  }
  __syncthreads();
  if (tid < BM) {
    atomicAdd(&rowsum [I0 + tid], srow[tid]);
    atomicAdd(&rowmask[I0 + tid], srowm[tid]);
    atomicAdd(&colsum [J0 + tid], scol[tid]);
    atomicAdd(&colmask[J0 + tid], scolm[tid]);
  }
}

// ---------------- row L2 norms of bf16 matrix (one wave per row) ----------------
__global__ __launch_bounds__(256)
void row_norm_kernel(const unsigned short* __restrict__ z, float* __restrict__ out, int rows)
{
  const int wg   = (blockIdx.x * blockDim.x + threadIdx.x) >> 5;
  const int lane = threadIdx.x & 31;
  if (wg >= rows) return;
  const unsigned short* p = z + (size_t)wg * HDIM;
  float s = 0.f;
  for (int d = lane; d < HDIM; d += 32) { const float x = bf2f(p[d]); s += x * x; }
#pragma unroll
  for (int off = 16; off; off >>= 1) s += __shfl_xor(s, off);
  if (lane == 0) out[wg] = sqrtf(s);
}

// ---------------- indexed pair dots + log-sigmoid accumulation ----------------
__global__ __launch_bounds__(256)
void pair_loss_kernel(const float* __restrict__ zma, const float* __restrict__ zsa,
                      const float* __restrict__ zmb, const float* __restrict__ zsb,
                      const int* __restrict__ ii, const int* __restrict__ jj,
                      int P, float sign, float* __restrict__ accum)
{
  const int wg   = (blockIdx.x * blockDim.x + threadIdx.x) >> 5;
  const int lane = threadIdx.x & 31;
  if (wg >= P) return;
  const int i = ii[wg], j = jj[wg];
  const float* am = zma + (size_t)i * HDIM;
  const float* as = zsa + (size_t)i * HDIM;
  const float* bm = zmb + (size_t)j * HDIM;
  const float* bs = zsb + (size_t)j * HDIM;
  float s = 0.f;
  for (int d = lane; d < HDIM; d += 32)
    s += (am[d] + as[d]) * (bm[d] + bs[d]);
#pragma unroll
  for (int off = 16; off; off >>= 1) s += __shfl_xor(s, off);
  if (lane == 0) {
    const float x  = sign * s;
    const float ls = fminf(x, 0.f) - log1pf(__expf(-fabsf(x)));  // log_sigmoid(x)
    atomicAdd(accum, ls);
  }
}

// ---------------- workspace zero ----------------
__global__ void zero_kernel(float* __restrict__ p, int n)
{
  const int i = blockIdx.x * blockDim.x + threadIdx.x;
  if (i < n) p[i] = 0.f;
}

// ---------------- final scalar assembly ----------------
__global__ __launch_bounds__(256)
void finalize_kernel(const float* __restrict__ red1, const float* __restrict__ red2,
                     const float* __restrict__ acc, int P, float* __restrict__ out)
{
  __shared__ float red[256];
  const int tid = threadIdx.x;
  float part[4] = {0.f, 0.f, 0.f, 0.f};
  for (int i = tid; i < NROWS; i += 256) {
    part[0] += -logf(red1[NROWS + i]     / (red1[i]             + 1e-8f));
    part[1] += -logf(red1[3 * NROWS + i] / (red1[2 * NROWS + i] + 1e-8f));
    part[2] += -logf(red2[NROWS + i]     / (red2[i]             + 1e-8f));
    part[3] += -logf(red2[3 * NROWS + i] / (red2[2 * NROWS + i] + 1e-8f));
  }
  float tot[4];
#pragma unroll
  for (int k = 0; k < 4; ++k) {
    red[tid] = part[k];
    __syncthreads();
    for (int off = 128; off; off >>= 1) {
      if (tid < off) red[tid] += red[tid + off];
      __syncthreads();
    }
    tot[k] = red[0];
    __syncthreads();
  }
  if (tid == 0) {
    const float inv = 1.0f / (float)NROWS;
    const float loss1 = LAM_F * tot[0] * inv + (1.f - LAM_F) * tot[1] * inv;
    const float loss2 = LAM_F * tot[2] * inv + (1.f - LAM_F) * tot[3] * inv;
    const float loss_main = (-acc[0] + acc[1]) / (float)P;
    out[0] = loss_main + loss1 + loss2;
  }
}

extern "C" void kernel_launch(void* const* d_in, const int* in_sizes, int n_in,
                              void* d_out, int out_size, void* d_ws, size_t ws_size,
                              hipStream_t stream)
{
  const float* z_mp1 = (const float*)d_in[0];
  const float* z_sc1 = (const float*)d_in[1];
  const float* pos1  = (const float*)d_in[2];
  const float* z_mp2 = (const float*)d_in[3];
  const float* z_sc2 = (const float*)d_in[4];
  const float* pos2  = (const float*)d_in[5];
  const float* W1    = (const float*)d_in[6];
  const float* b1    = (const float*)d_in[7];
  const float* W2    = (const float*)d_in[8];
  const float* b2    = (const float*)d_in[9];
  const int* pos_i   = (const int*)d_in[10];
  const int* pos_j   = (const int*)d_in[11];
  const int* neg_i   = (const int*)d_in[12];
  const int* neg_j   = (const int*)d_in[13];
  const int P  = in_sizes[10];
  const int Pn = in_sizes[12];

  // ---- workspace layout: bf16 region then f32 region ----
  const size_t WW  = (size_t)HDIM * HDIM;    // 262,144
  const size_t NHu = (size_t)NROWS * HDIM;   // 2,097,152
  unsigned short* wsu = (unsigned short*)d_ws;
  unsigned short* W1b = wsu;                 // [512 x 512] bf16
  unsigned short* W2b = wsu + WW;
  unsigned short* zb  = wsu + 2 * WW;        // [4096 x 512] bf16 input (reused)
  unsigned short* hb  = zb  + NHu;           // [4096 x 512] bf16 hidden
  unsigned short* zmb = hb  + NHu;           // [4096 x 512] bf16 proj(z_mp)
  unsigned short* zsb = zmb + NHu;           // [4096 x 512] bf16 proj(z_sc)
  float* fbase = (float*)(zsb + NHu);
  float* nm   = fbase;                       // [4096]
  float* ns   = nm + NROWS;                  // [4096]
  float* red1 = ns + NROWS;                  // 4 x 4096
  float* red2 = red1 + 4 * NROWS;            // 4 x 4096
  float* acc  = red2 + 4 * NROWS;            // [2]

  // zero reduction arrays + accumulators
  {
    const int nz = 8 * NROWS + 2;
    zero_kernel<<<(nz + 255) / 256, 256, 0, stream>>>(red1, nz);
  }

  // loss_main: indexed pair dots, log-sigmoid sums (f32 inputs directly)
  pair_loss_kernel<<<(P  + 7) / 8, 256, 0, stream>>>(z_mp1, z_sc1, z_mp2, z_sc2,
                                                     pos_i, pos_j, P,  1.0f, acc + 0);
  pair_loss_kernel<<<(Pn + 7) / 8, 256, 0, stream>>>(z_mp1, z_sc1, z_mp2, z_sc2,
                                                     neg_i, neg_j, Pn, -1.0f, acc + 1);

  const dim3 blk(256);
  const dim3 gproj(HDIM / BN, NROWS / BM);   // 4 x 32
  const dim3 gsim(NROWS / BN, NROWS / BM);   // 32 x 32
  const dim3 gnorm((NROWS * 32 + 255) / 256);
  const dim3 gcvtW(((int)(WW  >> 2) + 255) / 256);
  const dim3 gcvtZ(((int)(NHu >> 2) + 255) / 256);

  // weights -> bf16 once
  cvt_f32_bf16_kernel<<<gcvtW, blk, 0, stream>>>(W1, W1b, (int)WW);
  cvt_f32_bf16_kernel<<<gcvtW, blk, 0, stream>>>(W2, W2b, (int)WW);

  for (int l = 0; l < 2; ++l) {
    const float* zmp = l ? z_mp2 : z_mp1;
    const float* zsc = l ? z_sc2 : z_sc1;
    const float* pos = l ? pos2  : pos1;
    float* red       = l ? red2  : red1;

    // zm = proj(z_mp)
    cvt_f32_bf16_kernel<<<gcvtZ, blk, 0, stream>>>(zmp, zb, (int)NHu);
    gemm_bias_act_kernel<<<gproj, blk, 0, stream>>>(zb, W1b, b1, hb,  NROWS, HDIM, HDIM, 1);
    gemm_bias_act_kernel<<<gproj, blk, 0, stream>>>(hb, W2b, b2, zmb, NROWS, HDIM, HDIM, 0);
    // zs = proj(z_sc)
    cvt_f32_bf16_kernel<<<gcvtZ, blk, 0, stream>>>(zsc, zb, (int)NHu);
    gemm_bias_act_kernel<<<gproj, blk, 0, stream>>>(zb, W1b, b1, hb,  NROWS, HDIM, HDIM, 1);
    gemm_bias_act_kernel<<<gproj, blk, 0, stream>>>(hb, W2b, b2, zsb, NROWS, HDIM, HDIM, 0);

    row_norm_kernel<<<gnorm, blk, 0, stream>>>(zmb, nm, NROWS);
    row_norm_kernel<<<gnorm, blk, 0, stream>>>(zsb, ns, NROWS);

    sim_reduce_kernel<<<gsim, blk, 0, stream>>>(zmb, zsb, nm, ns, pos,
                                                red, red + NROWS,
                                                red + 2 * NROWS, red + 3 * NROWS,
                                                NROWS, HDIM);
  }

  finalize_kernel<<<1, blk, 0, stream>>>(red1, red2, acc, P, (float*)d_out);
}
